// FibonacciKANLayer_5299989643380
// MI455X (gfx1250) — matrix operand order repaired and checked by reading, not compiled, not run on hardware
//
#include <hip/hip_runtime.h>
#include <hip/hip_bf16.h>

typedef __attribute__((ext_vector_type(16))) _Float16 v16h;
typedef __attribute__((ext_vector_type(8)))  _Float16 v8h;
typedef __attribute__((ext_vector_type(4)))  _Float16 v4h;
typedef __attribute__((ext_vector_type(8)))  float    v8f;
typedef __attribute__((ext_vector_type(4)))  float    v4f;

#define BATCH     32768
#define IDIM      256
#define ODIM      256
#define NDEG      5                 // degrees 1..5 (F0 == 0 contributes nothing)
#define KCHUNK    32                // K per WMMA
#define NCHUNKS   (IDIM / KCHUNK)   // 8
#define NTILES_N  (ODIM / 16)       // 16
#define BM        64                // rows per block: x read exactly once
#define BN        256               // full output width per block

#if __has_builtin(__builtin_amdgcn_tanhf)
#define DEV_TANH(v) __builtin_amdgcn_tanhf(v)   // hardware V_TANH_F32
#else
#define DEV_TANH(v) tanhf(v)
#endif

static __device__ __forceinline__ v16h splat16(float s) {
    _Float16 h = (_Float16)s;
    v16h v;
#pragma unroll
    for (int i = 0; i < 16; ++i) v[i] = h;
    return v;
}

// ---------------------------------------------------------------------------
// Pack coeffs[i, o, d] (f32, stride-6 inner dim) into f16 WMMA-B fragments.
// Fragment block q = ((d-1)*NCHUNKS + ic)*NTILES_N + nt holds 32 lanes x 16
// halves contiguous; lane layout mirrors the ISA 16-bit B 32x16 layout:
//   n = nt*16 + lane%16 ; K_local = (half<8 ? half : half+8) + (lane<16 ? 0 : 8)
// ---------------------------------------------------------------------------
__global__ void fibkan_pack_coeffs(const float* __restrict__ coeffs,
                                   _Float16* __restrict__ bp) {
    int tid  = blockIdx.x * blockDim.x + threadIdx.x;  // one per packed half
    int q    = tid >> 9;
    int r    = tid & 511;
    int lane = r >> 4;
    int h    = r & 15;
    int nt   = q % NTILES_N;
    int ic   = (q / NTILES_N) % NCHUNKS;
    int d    = 1 + q / (NTILES_N * NCHUNKS);
    int kloc = ((h < 8) ? h : h + 8) + ((lane < 16) ? 0 : 8);
    int i    = ic * KCHUNK + kloc;
    int n    = nt * 16 + (lane & 15);
    bp[tid]  = (_Float16)coeffs[(i * ODIM + n) * (NDEG + 1) + d];
}

// ---------------------------------------------------------------------------
// Main kernel: y[b,o] = sum_i sum_{d=1..5} F_d(tanh(x[b,i])) * W[i,o,d]
// as a 16x16x32 f16 WMMA GEMM with basis expansion in registers.
// Block tile 64(M) x 256(N); 8 waves = 2(M) x 4(N); wave tile 2x4 16x16.
// ---------------------------------------------------------------------------
__global__ __launch_bounds__(256) void fibkan_wmma(
    const float* __restrict__ x,
    const _Float16* __restrict__ bp,
    float* __restrict__ out) {

    __shared__ _Float16 tileT[BM * IDIM];   // 32 KB: tanh(x) tile, f16

    const int tid = threadIdx.x;
    const int m0  = blockIdx.x * BM;

    // ---- Stage x tile -> hardware tanh -> f16 LDS (coalesced float4) ------
#pragma unroll
    for (int it = 0; it < (BM * IDIM) / (256 * 4); ++it) {
        int flat = it * (256 * 4) + tid * 4;
        v4f v = *(const v4f*)(x + (size_t)m0 * IDIM + flat);
        v4h hv;
        hv[0] = (_Float16)DEV_TANH(v[0]);
        hv[1] = (_Float16)DEV_TANH(v[1]);
        hv[2] = (_Float16)DEV_TANH(v[2]);
        hv[3] = (_Float16)DEV_TANH(v[3]);
        *(v4h*)(&tileT[flat]) = hv;
    }
    __syncthreads();

    // ---- Wave tiling -------------------------------------------------------
    const int wave  = tid >> 5;
    const int lane  = tid & 31;
    const int mbase = (wave >> 2) * 2;   // first of 2 m-tiles
    const int nbase = (wave & 3) * 4;    // first of 4 n-tiles
    const int lrow  = lane & 15;
    const int khalf = (lane < 16) ? 0 : 8;   // A-matrix K interleave

    v8f acc[2][4];
#pragma unroll
    for (int j = 0; j < 2; ++j)
#pragma unroll
        for (int jn = 0; jn < 4; ++jn)
#pragma unroll
            for (int e = 0; e < 8; ++e) acc[j][jn][e] = 0.0f;

    const v16h one   = splat16(1.0f);
    const v16h two   = splat16(2.0f);
    const v16h three = splat16(3.0f);

    for (int ic = 0; ic < NCHUNKS; ++ic) {
        // Gather A t-fragments from LDS per the 16-bit 16x32 A layout:
        // lane<16 holds K {0..7,16..23}, lane>=16 holds K {8..15,24..31}.
        v16h tf[2], t2[2];
#pragma unroll
        for (int j = 0; j < 2; ++j) {
            int m   = (mbase + j) * 16 + lrow;
            int off = m * IDIM + ic * KCHUNK + khalf;
            v8h lo = *(const v8h*)(&tileT[off]);
            v8h hi = *(const v8h*)(&tileT[off + 16]);
            tf[j] = __builtin_shufflevector(lo, hi,
                     0,1,2,3,4,5,6,7,8,9,10,11,12,13,14,15);
            t2[j] = tf[j] * tf[j];
        }

#pragma unroll
        for (int d = 1; d <= NDEG; ++d) {
            // B fragments: one contiguous 32B load per lane (L2 resident)
            v16h bf[4];
#pragma unroll
            for (int jn = 0; jn < 4; ++jn) {
                int q = ((d - 1) * NCHUNKS + ic) * NTILES_N + (nbase + jn);
                bf[jn] = *(const v16h*)(bp + (size_t)q * 512 + lane * 16);
            }
#pragma unroll
            for (int j = 0; j < 2; ++j) {
                v16h a;
                if      (d == 1) a = one;                          // F1 = 1
                else if (d == 2) a = tf[j];                        // F2 = t
                else if (d == 3) a = t2[j] + one;                  // F3 = t^2+1
                else if (d == 4) a = tf[j] * (t2[j] + two);        // F4 = t^3+2t
                else             a = t2[j] * (t2[j] + three) + one;// F5
#pragma unroll
                for (int jn = 0; jn < 4; ++jn) {
                    acc[j][jn] = __builtin_amdgcn_wmma_f32_16x16x32_f16(
                        false, a, false, bf[jn], (short)0, acc[j][jn],
                        false, false);
                }
            }
        }
    }

    // ---- Epilogue: C layout VGPR r -> M = r + (lane<16?0:8), N = lane%16 ---
    const int mo = (lane < 16) ? 0 : 8;
#pragma unroll
    for (int j = 0; j < 2; ++j) {
#pragma unroll
        for (int jn = 0; jn < 4; ++jn) {
            int m = m0 + (mbase + j) * 16 + mo;
            int n = (nbase + jn) * 16 + lrow;
#pragma unroll
            for (int r = 0; r < 8; ++r)
                out[(size_t)(m + r) * ODIM + n] = acc[j][jn][r];
        }
    }
}

extern "C" void kernel_launch(void* const* d_in, const int* in_sizes, int n_in,
                              void* d_out, int out_size, void* d_ws, size_t ws_size,
                              hipStream_t stream) {
    const float* x      = (const float*)d_in[0];   // [32768, 256] f32
    const float* coeffs = (const float*)d_in[1];   // [256, 256, 6] f32
    float* out          = (float*)d_out;           // [32768, 256] f32
    _Float16* bp        = (_Float16*)d_ws;         // 640 KB packed f16 B

    const int packN = NDEG * NCHUNKS * NTILES_N * 512;   // 327680 halves
    fibkan_pack_coeffs<<<packN / 256, 256, 0, stream>>>(coeffs, bp);

    fibkan_wmma<<<BATCH / BM, 256, 0, stream>>>(x, bp, out);   // 512 blocks
}
